// SigmaGPT_57226144252724
// MI455X (gfx1250) — matrix operand order repaired and checked by reading, not compiled, not run on hardware
//
#include <hip/hip_runtime.h>
#include <hip/hip_bf16.h>

typedef __attribute__((ext_vector_type(16))) _Float16 v16h;
typedef __attribute__((ext_vector_type(8)))  _Float16 h8;
typedef __attribute__((ext_vector_type(8)))  float    v8f;

static constexpr int BATCH = 2;
static constexpr int TSEQ  = 2048;
static constexpr int DM    = 1024;   // d_model
static constexpr int NH    = 16;     // heads
static constexpr int DHD   = 64;     // head dim (qk and v)
static constexpr int DHID  = 4096;   // ffn hidden
static constexpr int NV    = 32000;  // vocab
static constexpr int NL    = 2;      // layers
static constexpr int MR    = BATCH * TSEQ; // 4096 rows

__device__ __forceinline__ v16h cat16(h8 lo, h8 hi) {
  return __builtin_shufflevector(lo, hi, 0,1,2,3,4,5,6,7,8,9,10,11,12,13,14,15);
}

// ---------------- Tensor Data Mover (async tile DMA global->LDS) ----------------
#if defined(__has_builtin)
#  if __has_builtin(__builtin_amdgcn_tensor_load_to_lds) && \
      __has_builtin(__builtin_amdgcn_s_wait_tensorcnt)
#    define GEMM_USE_TDM 1
#  endif
#endif
#ifndef GEMM_USE_TDM
#  define GEMM_USE_TDM 0
#endif

#if GEMM_USE_TDM
typedef unsigned int tdm_u32x4 __attribute__((ext_vector_type(4)));
typedef int          tdm_i32x8 __attribute__((ext_vector_type(8)));
typedef int          tdm_i32x4 __attribute__((ext_vector_type(4)));

// DMA a 32(halves) x 128(rows) f16 tile, row stride `stride_halfs`, into LDS at
// byte offset `lds_byte_off`, with HW padding of 8 halves per 32-half row so the
// LDS image has pitch 40 halves (matches the manual staging layout exactly).
// D# encoding per CDNA5 ISA 8.3/8.4 (count=1, type=2, data_size=2B, 2D tile).
__device__ __forceinline__ void tdm_load_tile_f16(const _Float16* gsrc,
                                                  unsigned int lds_byte_off,
                                                  int stride_halfs)
{
  const unsigned long long ga = (unsigned long long)(size_t)gsrc;
  const unsigned int td0 = (unsigned int)stride_halfs;   // tensor_dim0 >= tile_dim0
  tdm_u32x4 g0;
  g0.x = 1u;                                             // count=1 (valid user D#)
  g0.y = lds_byte_off;                                   // lds_addr
  g0.z = (unsigned int)ga;                               // global_addr[31:0]
  g0.w = (unsigned int)((ga >> 32) & 0x01ffffffu) | 0x80000000u; // addr[56:32], type=2
  tdm_i32x8 g1;
  g1[0] = (1 << 16)        // data_size = 2 bytes
        | (1 << 20)        // pad_enable
        | (3 << 22)        // pad_interval: 16 DWORDs = one 32-half row
        | (3 << 25);       // pad_amount: 4 DWORDs = 8 halves
  g1[1] = (int)((td0 & 0xffffu) << 16);                  // tensor_dim0[15:0]
  g1[2] = (int)(td0 >> 16);                              // tensor_dim0[31:16]; tensor_dim1[15:0]=0
  g1[3] = (int)(16u | (32u << 16));                      // tensor_dim1[31:16]=16 (dim1=1<<20); tile_dim0=32
  g1[4] = 128;                                           // tile_dim1=128 rows; tile_dim2=0
  g1[5] = (int)td0;                                      // tensor_dim0_stride[31:0]
  g1[6] = 0;                                             // stride hi / dim1 stride (unused for 2D)
  g1[7] = 0;
  const tdm_i32x4 z4 = {0, 0, 0, 0};
#if __clang_major__ >= 23
  const tdm_i32x8 z8 = {0, 0, 0, 0, 0, 0, 0, 0};
  __builtin_amdgcn_tensor_load_to_lds(g0, g1, z4, z4, z8, 0);
#else
  __builtin_amdgcn_tensor_load_to_lds(g0, g1, z4, z4, 0);
#endif
}
#endif // GEMM_USE_TDM

// ---------------- embedding + double positional encoding ----------------
__global__ void embed_pe_kernel(const int* __restrict__ tokens,
                                const int* __restrict__ order,
                                const float* __restrict__ emb,
                                float* __restrict__ x)
{
  const int row = blockIdx.x;                 // b*T + t
  const int t   = row & (TSEQ - 1);
  const int tok = tokens[row];
  const int o_out = order[row] + 1;
  const int o_in  = (t == 0) ? 0 : (order[row - 1] + 1);
  const float* er = emb + (size_t)tok * DM;
  float* xr = x + (size_t)row * DM;
  for (int c = threadIdx.x; c < DM; c += 256) {
    const int j = (c < DM / 2) ? c : (c - DM / 2);
    const int o = (c < DM / 2) ? o_in : o_out;
    const float kf = (float)(j & 1);
    const float ex = (float)(j - (j & 1)) * (1.0f / (float)DM);
    const float freq = __powf(100000.0f, -ex);
    const float pe = __sinf((float)(o - 1) * freq + 1.57079632679f * kf);
    xr[c] = er[c] + pe;
  }
}

// ---------------- layernorm (f32 in -> f16 out) ----------------
__global__ void layernorm_f16_kernel(const float* __restrict__ x,
                                     const float* __restrict__ w,
                                     const float* __restrict__ b,
                                     _Float16* __restrict__ out)
{
  __shared__ float rs[256];
  __shared__ float rq[256];
  const int row = blockIdx.x;
  const int tid = threadIdx.x;
  const float* xr = x + (size_t)row * DM;
  float s = 0.0f, q = 0.0f;
  for (int c = tid; c < DM; c += 256) { float v = xr[c]; s += v; q += v * v; }
  rs[tid] = s; rq[tid] = q;
  __syncthreads();
  for (int off = 128; off > 0; off >>= 1) {
    if (tid < off) { rs[tid] += rs[tid + off]; rq[tid] += rq[tid + off]; }
    __syncthreads();
  }
  const float mean = rs[0] * (1.0f / DM);
  const float var  = rq[0] * (1.0f / DM) - mean * mean;
  const float rstd = rsqrtf(var + 1e-5f);
  _Float16* orow = out + (size_t)row * DM;
  for (int c = tid; c < DM; c += 256)
    orow[c] = (_Float16)((xr[c] - mean) * rstd * w[c] + b[c]);
}

// ---------------- f32 -> f16 convert ----------------
__global__ void cvt_f32_to_f16_kernel(const float* __restrict__ in,
                                      _Float16* __restrict__ out, int n)
{
  const int i = blockIdx.x * blockDim.x + threadIdx.x;
  if (i < n) out[i] = (_Float16)in[i];
}

// ---------------- WMMA GEMM: C[M,N] = A[M,K] * B, B from weight W ----------------
// TRANSB=true : W is [N,K] row-major (B[k][n] = W[n*K+k])  -- w_q/w_k/w_v/fc1/fc2/ro
// TRANSB=false: W is [K,N] row-major (B[k][n] = W[k*N+n])  -- w_o
enum { EPI_F16 = 0, EPI_RELU_BIAS_F16 = 1, EPI_RESID_F32 = 2, EPI_BIAS_F32 = 3 };

template<int EPI, bool TRANSB>
__launch_bounds__(256)
__global__ void gemm_wmma_kernel(const _Float16* __restrict__ A,
                                 const _Float16* __restrict__ W,
                                 const float* __restrict__ bias,
                                 void* __restrict__ outp,
                                 int M, int N, int K, float alpha)
{
  __shared__ __align__(16) _Float16 At[2][128 * 40];  // [m][k], pitch 40 halves
  __shared__ __align__(16) _Float16 Bt[2][128 * 40];  // [n][k], pitch 40 halves
  const int tid  = threadIdx.x;
  const int lane = tid & 31, wave = tid >> 5;
  const int wm = wave & 3, wn = wave >> 2;            // 4 x 2 wave grid
  const int half = lane >> 4, r = lane & 15;
  const int m0 = blockIdx.y * 128, n0 = blockIdx.x * 128;
  const v8f vzero = {0,0,0,0,0,0,0,0};
  v8f acc[2][4];
#pragma unroll
  for (int i = 0; i < 2; ++i)
#pragma unroll
    for (int j = 0; j < 4; ++j) acc[i][j] = vzero;

  auto compute_tile = [&](const _Float16* Ab, const _Float16* Bb) {
#pragma unroll
    for (int i = 0; i < 2; ++i) {
      // A fragment: lane row r, K = half*16 + 0..15 (contiguous)
      const _Float16* ar = &Ab[(wm * 32 + i * 16 + r) * 40];
      h8 alo = *(const h8*)&ar[half * 16];
      h8 ahi = *(const h8*)&ar[half * 16 + 8];
      v16h a = cat16(alo, ahi);
#pragma unroll
      for (int j = 0; j < 4; ++j) {
        // B fragment: lane col r, K = half*16 + 0..15 (contiguous)
        const _Float16* br = &Bb[(wn * 64 + j * 16 + r) * 40 + half * 16];
        h8 blo = *(const h8*)&br[0];
        h8 bhi = *(const h8*)&br[8];
        v16h bb = cat16(blo, bhi);
        acc[i][j] = __builtin_amdgcn_wmma_f32_16x16x32_f16(
            false, a, false, bb, (short)0, acc[i][j], false, false);
      }
    }
  };

#if GEMM_USE_TDM
  if constexpr (TRANSB) {
    // Double-buffered TDM pipeline: DMA next A/B tiles while WMMA runs.
    const unsigned int aoff0 = (unsigned int)(size_t)&At[0][0];
    const unsigned int boff0 = (unsigned int)(size_t)&Bt[0][0];
    const unsigned int bufbytes = 128u * 40u * 2u;
    if (wave == 0) tdm_load_tile_f16(&A[(size_t)m0 * K], aoff0, K);
    if (wave == 1) tdm_load_tile_f16(&W[(size_t)n0 * K], boff0, K);
    if (wave < 2) __builtin_amdgcn_s_wait_tensorcnt(0);
    __syncthreads();
    for (int k0 = 0; k0 < K; k0 += 32) {
      const int cur = (k0 >> 5) & 1;
      if (k0 + 32 < K) {
        const unsigned int nb = (unsigned int)(1 - cur) * bufbytes;
        if (wave == 0)
          tdm_load_tile_f16(&A[(size_t)m0 * K + k0 + 32], aoff0 + nb, K);
        if (wave == 1)
          tdm_load_tile_f16(&W[(size_t)n0 * K + k0 + 32], boff0 + nb, K);
      }
      compute_tile(&At[cur][0], &Bt[cur][0]);
      if (wave < 2 && k0 + 32 < K) __builtin_amdgcn_s_wait_tensorcnt(0);
      __syncthreads();
    }
  } else
#endif
  {
    for (int k0 = 0; k0 < K; k0 += 32) {
      __syncthreads();
      // stage A tile (128x32)
#pragma unroll
      for (int c = 0; c < 2; ++c) {
        const int chunk = tid * 2 + c;               // 512 chunks of 8 halves
        const int mm = chunk >> 2, kk = (chunk & 3) << 3;
        *(uint4*)&At[0][mm * 40 + kk] =
            *(const uint4*)&A[(size_t)(m0 + mm) * K + k0 + kk];
      }
      // stage B tile as [n][k]
      if (TRANSB) {
#pragma unroll
        for (int c = 0; c < 2; ++c) {
          const int chunk = tid * 2 + c;
          const int nn = chunk >> 2, kk = (chunk & 3) << 3;
          *(uint4*)&Bt[0][nn * 40 + kk] =
              *(const uint4*)&W[(size_t)(n0 + nn) * K + k0 + kk];
        }
      } else {
#pragma unroll
        for (int c = 0; c < 2; ++c) {
          const int chunk = tid * 2 + c;             // 32 k-rows x 16 n-chunks
          const int kk = chunk >> 4, nn = (chunk & 15) << 3;
          h8 wv = *(const h8*)&W[(size_t)(k0 + kk) * N + n0 + nn];
#pragma unroll
          for (int j = 0; j < 8; ++j) Bt[0][(nn + j) * 40 + kk] = wv[j];
        }
      }
      if (k0 + 32 < K) {
        __builtin_prefetch(&A[(size_t)(m0 + (tid >> 1)) * K + k0 + 32], 0, 0);
        if (TRANSB)
          __builtin_prefetch(&W[(size_t)(n0 + (tid >> 1)) * K + k0 + 32], 0, 0);
      }
      __syncthreads();
      compute_tile(&At[0][0], &Bt[0][0]);
    }
  }

  // epilogue: C element e -> m = half*8+e, n = lane%16
#pragma unroll
  for (int i = 0; i < 2; ++i) {
#pragma unroll
    for (int j = 0; j < 4; ++j) {
#pragma unroll
      for (int e = 0; e < 8; ++e) {
        const int m = m0 + wm * 32 + i * 16 + half * 8 + e;
        const int n = n0 + wn * 64 + j * 16 + r;
        const size_t idx = (size_t)m * N + n;
        const float v = acc[i][j][e];
        if (EPI == EPI_F16) {
          ((_Float16*)outp)[idx] = (_Float16)(v * alpha);
        } else if (EPI == EPI_RELU_BIAS_F16) {
          const float t = v + bias[n];
          ((_Float16*)outp)[idx] = (_Float16)(t > 0.0f ? t : 0.0f);
        } else if (EPI == EPI_RESID_F32) {
          const float t = v + (bias ? bias[n] : 0.0f);
          ((float*)outp)[idx] += t;
        } else { // EPI_BIAS_F32
          ((float*)outp)[idx] = v + bias[n];
        }
      }
    }
  }
}

// ---------------- flash attention with WMMA (causal) ----------------
// Q/K/V layout: [b*T + t][h*64 + d] (row stride DM). Q pre-scaled by 1/8.
__launch_bounds__(128)
__global__ void attn_kernel(const _Float16* __restrict__ Qm,
                            const _Float16* __restrict__ Km,
                            const _Float16* __restrict__ Vm,
                            _Float16* __restrict__ Ym)
{
  __shared__ __align__(16) _Float16 Kt[32 * 72];      // [s][d]
  __shared__ __align__(16) _Float16 Vt[64 * 40];      // [d][s] (transposed)
  __shared__ __align__(16) _Float16 Pt[4][16 * 40];   // per-wave P tile [m][s]
  const float NEG = -3.0e38f;
  const int tid = threadIdx.x;
  const int lane = tid & 31, wave = tid >> 5;
  const int half = lane >> 4, r = lane & 15;
  const int b = blockIdx.z, h = blockIdx.y;
  const int q0 = blockIdx.x * 64;
  const int mrow = q0 + wave * 16;                    // wave's first query row
  const size_t base = ((size_t)b * TSEQ) * DM + (size_t)h * DHD;

  // Q as A fragments (K-dim 64 -> 2 frags), loaded straight from global
  const _Float16* qp = Qm + base + (size_t)(mrow + r) * DM;
  v16h qa[2];
#pragma unroll
  for (int f = 0; f < 2; ++f) {
    h8 lo = *(const h8*)&qp[f * 32 + half * 16];
    h8 hi = *(const h8*)&qp[f * 32 + half * 16 + 8];
    qa[f] = cat16(lo, hi);
  }

  float mrun[8], lrun[8];
#pragma unroll
  for (int e = 0; e < 8; ++e) { mrun[e] = NEG; lrun[e] = 0.0f; }
  const v8f vzero = {0,0,0,0,0,0,0,0};
  v8f o[4] = {vzero, vzero, vzero, vzero};

  const int nblk = (q0 + 64) >> 5;                    // causal: only s <= q block end
  for (int kb = 0; kb < nblk; ++kb) {
    const int s0 = kb << 5;
    __syncthreads();
    { // cooperative K/V tile load (32 keys x 64 dims)
      const int srow = tid >> 2;
      const int dc = (tid & 3) << 4;
      const _Float16* kp = Km + base + (size_t)(s0 + srow) * DM + dc;
      *(uint4*)&Kt[srow * 72 + dc]     = *(const uint4*)&kp[0];
      *(uint4*)&Kt[srow * 72 + dc + 8] = *(const uint4*)&kp[8];
      const _Float16* vp = Vm + base + (size_t)(s0 + srow) * DM + dc;
#pragma unroll
      for (int j = 0; j < 16; ++j) Vt[(dc + j) * 40 + srow] = vp[j];
    }
    __syncthreads();

    // S = Q K^T  (two 16-col subtiles, K-dim 64)
    v8f s[2] = {vzero, vzero};
#pragma unroll
    for (int j = 0; j < 2; ++j) {
#pragma unroll
      for (int f = 0; f < 2; ++f) {
        const _Float16* kr = &Kt[(j * 16 + r) * 72 + f * 32 + half * 16];
        h8 lo = *(const h8*)&kr[0];
        h8 hi = *(const h8*)&kr[8];
        v16h bk = cat16(lo, hi);
        s[j] = __builtin_amdgcn_wmma_f32_16x16x32_f16(
            false, qa[f], false, bk, (short)0, s[j], false, false);
      }
    }

    // causal mask + online softmax stats
    float pj[2][8], pm[8];
#pragma unroll
    for (int e = 0; e < 8; ++e) {
      const int m = mrow + half * 8 + e;
#pragma unroll
      for (int j = 0; j < 2; ++j) {
        float sv = s[j][e];
        if (s0 + j * 16 + r > m) sv = NEG;
        pj[j][e] = sv;
      }
      pm[e] = fmaxf(pj[0][e], pj[1][e]);
    }
#pragma unroll
    for (int off = 1; off < 16; off <<= 1) {
#pragma unroll
      for (int e = 0; e < 8; ++e) pm[e] = fmaxf(pm[e], __shfl_xor(pm[e], off, 32));
    }
    float al[8], ps[8];
#pragma unroll
    for (int e = 0; e < 8; ++e) {
      const float mn = fmaxf(mrun[e], pm[e]);
      float a = __expf(mrun[e] - mn);
      if (mrun[e] <= NEG) a = 0.0f;
      al[e] = a; mrun[e] = mn;
      float p0 = __expf(pj[0][e] - mn); if (pj[0][e] <= NEG) p0 = 0.0f;
      float p1 = __expf(pj[1][e] - mn); if (pj[1][e] <= NEG) p1 = 0.0f;
      pj[0][e] = p0; pj[1][e] = p1; ps[e] = p0 + p1;
    }
#pragma unroll
    for (int off = 1; off < 16; off <<= 1) {
#pragma unroll
      for (int e = 0; e < 8; ++e) ps[e] += __shfl_xor(ps[e], off, 32);
    }
#pragma unroll
    for (int e = 0; e < 8; ++e) lrun[e] = lrun[e] * al[e] + ps[e];
#pragma unroll
    for (int dj = 0; dj < 4; ++dj)
#pragma unroll
      for (int e = 0; e < 8; ++e) o[dj][e] *= al[e];

    // P (C layout) -> per-wave LDS -> reload as A fragment
    _Float16* pw = &Pt[wave][0];
#pragma unroll
    for (int e = 0; e < 8; ++e)
#pragma unroll
      for (int j = 0; j < 2; ++j)
        pw[(half * 8 + e) * 40 + j * 16 + r] = (_Float16)pj[j][e];
    v16h pa;
    {
      const _Float16* pr = &pw[r * 40];
      h8 lo = *(const h8*)&pr[half * 16];
      h8 hi = *(const h8*)&pr[half * 16 + 8];
      pa = cat16(lo, hi);
    }
    // O += P @ V  (4 d-subtiles, K-dim 32)
#pragma unroll
    for (int dj = 0; dj < 4; ++dj) {
      const _Float16* vr = &Vt[(dj * 16 + r) * 40 + half * 16];
      h8 lo = *(const h8*)&vr[0];
      h8 hi = *(const h8*)&vr[8];
      v16h bv = cat16(lo, hi);
      o[dj] = __builtin_amdgcn_wmma_f32_16x16x32_f16(
          false, pa, false, bv, (short)0, o[dj], false, false);
    }
  }

  // normalize and store y[b, t, h*64 + d] as f16
#pragma unroll
  for (int e = 0; e < 8; ++e) {
    const int m = mrow + half * 8 + e;
    const float inv = 1.0f / lrun[e];
    _Float16* yr = Ym + ((size_t)(b * TSEQ + m)) * DM + (size_t)h * DHD;
#pragma unroll
    for (int dj = 0; dj < 4; ++dj)
      yr[dj * 16 + r] = (_Float16)(o[dj][e] * inv);
  }
}

// ---------------- host orchestration ----------------
extern "C" void kernel_launch(void* const* d_in, const int* in_sizes, int n_in,
                              void* d_out, int out_size, void* d_ws, size_t ws_size,
                              hipStream_t stream)
{
  (void)in_sizes; (void)n_in; (void)out_size; (void)ws_size;
  const int*   tokens = (const int*)d_in[0];
  const int*   order  = (const int*)d_in[1];
  const float* emb    = (const float*)d_in[2];
  const float* ln1_w  = (const float*)d_in[3];
  const float* ln1_b  = (const float*)d_in[4];
  const float* w_q    = (const float*)d_in[5];
  const float* w_k    = (const float*)d_in[6];
  const float* w_v    = (const float*)d_in[7];
  const float* w_o    = (const float*)d_in[8];
  const float* ln2_w  = (const float*)d_in[9];
  const float* ln2_b  = (const float*)d_in[10];
  const float* fc1_w  = (const float*)d_in[11];
  const float* fc1_b  = (const float*)d_in[12];
  const float* fc2_w  = (const float*)d_in[13];
  const float* fc2_b  = (const float*)d_in[14];
  const float* ro_w   = (const float*)d_in[15];
  const float* ro_b   = (const float*)d_in[16];
  float* out = (float*)d_out;

  char* wsb = (char*)d_ws;
  size_t off = 0;
  auto take = [&](size_t bytes) -> void* {
    void* r = wsb + off;
    off += (bytes + 255) & ~(size_t)255;
    return r;
  };
  float*    x   = (float*)   take((size_t)MR * DM * sizeof(float));
  _Float16* xn  = (_Float16*)take((size_t)MR * DM * sizeof(_Float16));
  _Float16* q16 = (_Float16*)take((size_t)MR * DM * sizeof(_Float16));
  _Float16* k16 = (_Float16*)take((size_t)MR * DM * sizeof(_Float16));
  _Float16* v16 = (_Float16*)take((size_t)MR * DM * sizeof(_Float16));
  _Float16* y16 = (_Float16*)take((size_t)MR * DM * sizeof(_Float16));
  _Float16* h16 = (_Float16*)take((size_t)MR * DHID * sizeof(_Float16));
  _Float16* wb  = (_Float16*)take((size_t)NV * DM * sizeof(_Float16));

  const dim3 blk256(256), blk128(128);
  auto cvt = [&](const float* src, _Float16* dst, int n) {
    cvt_f32_to_f16_kernel<<<(n + 255) / 256, blk256, 0, stream>>>(src, dst, n);
  };

  // x = emb[tokens] + double_pe(order)
  embed_pe_kernel<<<MR, blk256, 0, stream>>>(tokens, order, emb, x);

  for (int l = 0; l < NL; ++l) {
    // ---- attention block ----
    layernorm_f16_kernel<<<MR, blk256, 0, stream>>>(x, ln1_w + l * DM, ln1_b + l * DM, xn);
    // Q (scaled by 1/sqrt(64)=0.125), K, V projections (all [4096,1024]x[1024,1024])
    cvt(w_q + (size_t)l * DM * DM, wb, DM * DM);
    gemm_wmma_kernel<EPI_F16, true><<<dim3(DM / 128, MR / 128), blk256, 0, stream>>>(
        xn, wb, nullptr, q16, MR, DM, DM, 0.125f);
    cvt(w_k + (size_t)l * DM * DM, wb, DM * DM);
    gemm_wmma_kernel<EPI_F16, true><<<dim3(DM / 128, MR / 128), blk256, 0, stream>>>(
        xn, wb, nullptr, k16, MR, DM, DM, 1.0f);
    cvt(w_v + (size_t)l * DM * DM, wb, DM * DM);
    gemm_wmma_kernel<EPI_F16, true><<<dim3(DM / 128, MR / 128), blk256, 0, stream>>>(
        xn, wb, nullptr, v16, MR, DM, DM, 1.0f);
    // flash attention
    attn_kernel<<<dim3(TSEQ / 64, NH, BATCH), blk128, 0, stream>>>(q16, k16, v16, y16);
    // x += y @ w_o  (w_o stored [K,N] -> non-transposed B, manual staging path)
    cvt(w_o + (size_t)l * DM * DM, wb, DM * DM);
    gemm_wmma_kernel<EPI_RESID_F32, false><<<dim3(DM / 128, MR / 128), blk256, 0, stream>>>(
        y16, wb, nullptr, x, MR, DM, DM, 1.0f);

    // ---- FFN block ----
    layernorm_f16_kernel<<<MR, blk256, 0, stream>>>(x, ln2_w + l * DM, ln2_b + l * DM, xn);
    cvt(fc1_w + (size_t)l * DHID * DM, wb, DHID * DM);
    gemm_wmma_kernel<EPI_RELU_BIAS_F16, true><<<dim3(DHID / 128, MR / 128), blk256, 0, stream>>>(
        xn, wb, fc1_b + l * DHID, h16, MR, DHID, DM, 1.0f);
    cvt(fc2_w + (size_t)l * DM * DHID, wb, DM * DHID);
    gemm_wmma_kernel<EPI_RESID_F32, true><<<dim3(DM / 128, MR / 128), blk256, 0, stream>>>(
        h16, wb, fc2_b + l * DM, x, MR, DM, DHID, 1.0f);
  }

  // ---- readout: logits = x @ ro_w.T + ro_b ----
  cvt(x, xn, MR * DM);
  cvt(ro_w, wb, NV * DM);
  gemm_wmma_kernel<EPI_BIAS_F32, true><<<dim3(NV / 128, MR / 128), blk256, 0, stream>>>(
      xn, wb, ro_b, out, MR, NV, DM, 1.0f);
}
